// SlowLSTM_20315195310597
// MI455X (gfx1250) — compile-verified, matrix operand order checked by reading, not run on hardware
//
#include <hip/hip_runtime.h>
#include <hip/hip_bf16.h>
#include <stdint.h>
#include <stddef.h>

// ---------------------------------------------------------------------------
// LSTM cell, B=4096, IN=H=2048.  gates = [x|h] @ [Wx;Wh] + b  -> fused epilogue.
// bf16 WMMA (v_wmma_f32_16x16x32_bf16) with f32 accumulate; async global->LDS
// staging when the gfx1250 async builtins are available.
// ---------------------------------------------------------------------------

typedef __attribute__((ext_vector_type(16))) __bf16 v16bf;
typedef __attribute__((ext_vector_type(8)))  __bf16 v8bf;
typedef __attribute__((ext_vector_type(8)))  float  v8f;
typedef __attribute__((vector_size(16)))     int    vi4;   // 4 x i32 (b128)

#define AS1 __attribute__((address_space(1)))
#define AS3 __attribute__((address_space(3)))

#if defined(__HIP_DEVICE_COMPILE__) && \
    __has_builtin(__builtin_amdgcn_global_load_async_to_lds_b128) && \
    __has_builtin(__builtin_amdgcn_s_wait_asynccnt)
#define HAVE_ASYNC 1
#else
#define HAVE_ASYNC 0
#endif

__device__ __forceinline__ void copy16(const uint16_t* g, uint16_t* l) {
#if HAVE_ASYNC
  // global_load_async_to_lds_b128: per-lane 16B global -> LDS, tracked by ASYNCcnt.
  // Flat LDS address low 32 bits == LDS byte address (ISA 10.2 aperture rules).
  __builtin_amdgcn_global_load_async_to_lds_b128(
      (AS1 vi4*)(uintptr_t)g,
      (AS3 vi4*)(uint32_t)(uintptr_t)l,
      0, 0);
#else
  *(uint4*)l = *(const uint4*)g;   // global_load_b128 + ds_store_b128 fallback
#endif
}

template <int N>
__device__ __forceinline__ void wait_async() {
#if HAVE_ASYNC
  __builtin_amdgcn_s_wait_asynccnt(N);
#endif
}

__device__ __forceinline__ uint32_t f2bf(float f) {   // RNE f32 -> bf16 bits
  uint32_t u = __float_as_uint(f);
  return (u + 0x7FFFu + ((u >> 16) & 1u)) >> 16;
}

__device__ __forceinline__ float sigmoidf_fast(float x) {
  return 1.0f / (1.0f + __expf(-x));                  // v_exp_f32 (TRANS pipe)
}

// ---------------------------------------------------------------------------
// Kernel 1: A_bf16[4096][4096] = [x | h] converted to bf16 (4 floats/thread).
// ---------------------------------------------------------------------------
__global__ __launch_bounds__(256)
void pack_A(const float* __restrict__ x, const float* __restrict__ h,
            uint16_t* __restrict__ A) {
  size_t i = (size_t)blockIdx.x * blockDim.x + threadIdx.x;  // one per 4 elems
  int col4 = (int)(i & 1023);          // 4096/4 chunks per row
  int r    = (int)(i >> 10);
  int k    = col4 << 2;
  const float* src = (k < 2048) ? (x + (size_t)r * 2048 + k)
                                : (h + (size_t)r * 2048 + (k - 2048));
  float4 v = *(const float4*)src;
  uint2 o;
  o.x = f2bf(v.x) | (f2bf(v.y) << 16);
  o.y = f2bf(v.z) | (f2bf(v.w) << 16);
  *(uint2*)(A + (size_t)r * 4096 + k) = o;
}

// ---------------------------------------------------------------------------
// Kernel 2: Bt_bf16[8192][4096] = transpose of [Wx;Wh] (gate-major in N),
// via 32x32 LDS tile so both read and write are coalesced.
// ---------------------------------------------------------------------------
__global__ __launch_bounds__(256)
void pack_Bt(const float* __restrict__ wxi, const float* __restrict__ wxf,
             const float* __restrict__ wxo, const float* __restrict__ wxc,
             const float* __restrict__ whi, const float* __restrict__ whf,
             const float* __restrict__ who, const float* __restrict__ whc,
             uint16_t* __restrict__ Bt) {
  __shared__ uint16_t tile[32][33];
  const float* wx[4] = {wxi, wxf, wxo, wxc};
  const float* wh[4] = {whi, whf, who, whc};

  int kt = blockIdx.x * 32;            // K tile base (0..4095)
  int nt = blockIdx.y * 32;            // N tile base (0..8191)
  int g     = nt >> 11;                // gate (2048 cols per gate)
  int hcol0 = nt & 2047;
  const float* W = (kt < 2048) ? wx[g] : wh[g];
  int kk0 = kt & 2047;

  int tx = threadIdx.x, ty = threadIdx.y;   // (32, 8)
#pragma unroll
  for (int j = 0; j < 4; ++j) {
    int kr = ty + j * 8;
    tile[kr][tx] = (uint16_t)f2bf(W[(size_t)(kk0 + kr) * 2048 + hcol0 + tx]);
  }
  __syncthreads();
#pragma unroll
  for (int j = 0; j < 4; ++j) {
    int nr = ty + j * 8;
    Bt[(size_t)(nt + nr) * 4096 + kt + tx] = tile[tx][nr];
  }
}

// ---------------------------------------------------------------------------
// Kernel 3: fused GEMM + LSTM epilogue.
//   block tile: 128 batch rows x (32 h-cols x 4 gates); 8 waves (wave32)
//   wave tile : 32 rows x 16 h-cols, all 4 gates -> 8 accumulators (v8f)
// ---------------------------------------------------------------------------
__global__ __launch_bounds__(256)
void lstm_wmma(const uint16_t* __restrict__ A, const uint16_t* __restrict__ Bt,
               const float* __restrict__ cprev,
               const float* __restrict__ bi, const float* __restrict__ bf,
               const float* __restrict__ bo, const float* __restrict__ bc,
               float* __restrict__ outh, float* __restrict__ outc) {
  constexpr int K  = 4096, Hd = 2048;
  constexpr int BM = 128, BK = 32, LDK = BK + 8;   // pad keeps 16B alignment
  constexpr int NK = K / BK;

  __shared__ uint16_t sA[2][BM][LDK];
  __shared__ uint16_t sB[2][BM][LDK];

  const int tid  = threadIdx.x;
  const int lane = tid & 31;
  const int wave = tid >> 5;
  const int mW   = wave & 3;           // 4 waves along M, 32 rows each
  const int nW   = wave >> 2;          // 2 waves along N, 16 cols each

  const int r0 = blockIdx.x * BM;      // batch-row base
  const int n0 = blockIdx.y * 32;      // h-col base

  v8f acc[4][2] = {};                  // [gate][m-frag]

  auto stage = [&](int buf, int k0) {
#pragma unroll
    for (int c = 0; c < 2; ++c) {      // 512 16B chunks per tile / 256 threads
      int chunk = tid + c * 256;
      int row = chunk >> 2;            // 0..127
      int kc  = (chunk & 3) * 8;       // 0,8,16,24
      copy16(A + (size_t)(r0 + row) * K + k0 + kc, &sA[buf][row][kc]);
      int gn = (row >> 5) * Hd + n0 + (row & 31);    // gate strip -> global N
      copy16(Bt + (size_t)gn * K + k0 + kc, &sB[buf][row][kc]);
    }
  };

  union BF16x16 { v16bf v; v8bf h[2]; };
  const int lr = lane & 15;            // row within 16x16 frag
  const int kh = (lane >> 4) << 4;     // K half: 0 or 16

  stage(0, 0);
  for (int kt = 0; kt < NK; ++kt) {
    int buf = kt & 1;
    if (kt + 1 < NK) {
      stage(buf ^ 1, (kt + 1) * BK);
      wait_async<4>();                 // this stage's 4 async ops complete
    } else {
      wait_async<0>();
    }
    __syncthreads();

    BF16x16 afr[2], bfr[4];
#pragma unroll
    for (int mi = 0; mi < 2; ++mi) {
      const uint16_t* p = &sA[buf][mW * 32 + mi * 16 + lr][kh];
      afr[mi].h[0] = *(const v8bf*)p;
      afr[mi].h[1] = *(const v8bf*)(p + 8);
    }
#pragma unroll
    for (int g = 0; g < 4; ++g) {
      const uint16_t* p = &sB[buf][g * 32 + nW * 16 + lr][kh];
      bfr[g].h[0] = *(const v8bf*)p;
      bfr[g].h[1] = *(const v8bf*)(p + 8);
    }
#pragma unroll
    for (int g = 0; g < 4; ++g)
#pragma unroll
      for (int mi = 0; mi < 2; ++mi)
        acc[g][mi] = __builtin_amdgcn_wmma_f32_16x16x32_bf16(
            false, afr[mi].v, false, bfr[g].v,
            (short)0, acc[g][mi], false, false);
    __syncthreads();
  }

  // Epilogue: C/D layout: VGPR r, lanes 0-15 -> M=r; lanes 16-31 -> M=r+8.
  const int col   = n0 + nW * 16 + (lane & 15);
  const float vbi = bi[col], vbf = bf[col], vbo = bo[col], vbc = bc[col];
  const int rbase = r0 + mW * 32 + ((lane >> 4) << 3);
#pragma unroll
  for (int mi = 0; mi < 2; ++mi) {
#pragma unroll
    for (int r = 0; r < 8; ++r) {
      int R = rbase + mi * 16 + r;
      float iv = sigmoidf_fast(acc[0][mi][r] + vbi);
      float fv = sigmoidf_fast(acc[1][mi][r] + vbf);
      float ov = sigmoidf_fast(acc[2][mi][r] + vbo);
      float gv = tanhf(acc[3][mi][r] + vbc);
      float cp = cprev[(size_t)R * Hd + col];
      float cn = cp * fv + iv * gv;
      float hn = ov * tanhf(cn);
      outh[(size_t)R * Hd + col] = hn;
      outc[(size_t)R * Hd + col] = cn;
    }
  }
}

// ---------------------------------------------------------------------------
extern "C" void kernel_launch(void* const* d_in, const int* in_sizes, int n_in,
                              void* d_out, int out_size, void* d_ws, size_t ws_size,
                              hipStream_t stream) {
  (void)in_sizes; (void)n_in; (void)out_size; (void)ws_size;
  const float* x    = (const float*)d_in[0];
  const float* h    = (const float*)d_in[1];
  const float* c    = (const float*)d_in[2];
  const float* w_xi = (const float*)d_in[3];
  const float* w_xf = (const float*)d_in[4];
  const float* w_xo = (const float*)d_in[5];
  const float* w_xc = (const float*)d_in[6];
  const float* w_hi = (const float*)d_in[7];
  const float* w_hf = (const float*)d_in[8];
  const float* w_ho = (const float*)d_in[9];
  const float* w_hc = (const float*)d_in[10];
  const float* b_i  = (const float*)d_in[11];
  const float* b_f  = (const float*)d_in[12];
  const float* b_o  = (const float*)d_in[13];
  const float* b_c  = (const float*)d_in[14];

  // Workspace: A_bf16 [4096x4096] (32MB) then Bt_bf16 [8192x4096] (64MB).
  uint16_t* A  = (uint16_t*)d_ws;
  uint16_t* Bt = (uint16_t*)((char*)d_ws + (size_t)4096 * 4096 * 2);

  float* outh = (float*)d_out;
  float* outc = outh + (size_t)4096 * 2048;

  pack_A<<<(4096 * 4096 / 4) / 256, 256, 0, stream>>>(x, h, A);
  pack_Bt<<<dim3(4096 / 32, 8192 / 32), dim3(32, 8), 0, stream>>>(
      w_xi, w_xf, w_xo, w_xc, w_hi, w_hf, w_ho, w_hc, Bt);
  lstm_wmma<<<dim3(4096 / 128, 2048 / 32), 256, 0, stream>>>(
      A, Bt, c, b_i, b_f, b_o, b_c, outh, outc);
}
